// AutoregressiveCDE_51951924412752
// MI455X (gfx1250) — compile-verified, matrix operand order checked by reading, not run on hardware
//
#include <hip/hip_runtime.h>
#include <hip/hip_bf16.h>
#include <stdint.h>

#define T_N   100
#define W_N   128
#define ROWS  16
#define NSUB  4

typedef __attribute__((ext_vector_type(16))) _Float16 v16h;
typedef __attribute__((ext_vector_type(8)))  float    v8f;

// softplus = ln2 * log2(1 + exp2(x*log2e)); raw v_exp_f32/v_log_f32, branchless guard
static __device__ __forceinline__ float sp_act(float x) {
    float r = 0.69314718056f *
              __builtin_amdgcn_logf(1.f + __builtin_amdgcn_exp2f(x * 1.44269504089f));
    return (x > 20.f) ? x : r;
}
// tanh = 1 - 2/(exp2(2*log2e*x)+1); exact at both saturation limits
static __device__ __forceinline__ float tanh_fast(float x) {
    float t = __builtin_amdgcn_exp2f(x * 2.88539008178f);
    return 1.f - 2.f * __builtin_amdgcn_rcpf(t + 1.f);
}

static __device__ __forceinline__ uint32_t pack2h(float a, float b) {
    union { _Float16 h[2]; uint32_t u; } v;
    v.h[0] = (_Float16)a; v.h[1] = (_Float16)b;
    return v.u;
}

__global__ __launch_bounds__(256)
void cde_persistent_kernel(const float* __restrict__ ts,
                           const float* __restrict__ ys,
                           const int*   __restrict__ ctrl_p,
                           const int*   __restrict__ train_p,
                           const float* __restrict__ w0,
                           const float* __restrict__ b0,
                           const float* __restrict__ w1,
                           const float* __restrict__ b1,
                           const float* __restrict__ wf,
                           const float* __restrict__ bf,
                           float* __restrict__ out,
                           int Bn)
{
    // ---- LDS: weights + activation staging only (~48 KB); ODE state is in registers
    __shared__ uint32_t Bpack[64 * W_N];     // w1^T f16 k-pairs (32 KB): WMMA-B, GEMM1
    __shared__ uint32_t Apack[ROWS * 64];    // H1  f16 k-pairs (4 KB):  WMMA-A, GEMM1
    __shared__ uint32_t h2pack[ROWS * 64];   // H2  f16 k-pairs (4 KB):  WMMA-A, GEMM2
    __shared__ uint32_t wfpack[64 * 16];     // wf^T f16 k-pairs, cols 9..15 zero (4 KB)
    __shared__ float    w0s[W_N * 3];
    __shared__ float    b0s[W_N];
    __shared__ float    sb1[W_N];
    __shared__ float    bfs[16];
    __shared__ float    tss[T_N];
    __shared__ float    Fbuf[ROWS][12];      // tanh(3x3) per row

    const int tid   = threadIdx.x;
    const int row0  = blockIdx.x * ROWS;
    const int ctrl  = ctrl_p[0];
    const int train = train_p[0];

    const int r      = tid >> 4;             // this thread's row (fixed)
    const int sg     = tid & 15;             // sub-slot within row
    const int row    = row0 + r;
    const bool rowok = row < Bn;
    const bool writer = (sg == 0);

    _Float16* h2h = reinterpret_cast<_Float16*>(h2pack);

    // ---- one-time weight staging ------------------------------------------
    for (int i = tid; i < W_N * 3; i += 256) w0s[i] = w0[i];
    for (int i = tid; i < W_N; i += 256) { b0s[i] = b0[i]; sb1[i] = b1[i]; }
    if (tid < 16) bfs[tid] = (tid < 9) ? bf[tid] : 0.f;
    for (int i = tid; i < T_N; i += 256) tss[i] = ts[i];
    for (int idx = tid; idx < 64 * W_N; idx += 256) {
        const int kk = idx >> 7, i = idx & 127;
        Bpack[kk * W_N + i] = pack2h(w1[i * W_N + 2 * kk], w1[i * W_N + 2 * kk + 1]);
    }
    for (int idx = tid; idx < 64 * 16; idx += 256) {
        const int kk = idx >> 4, p = idx & 15;
        wfpack[kk * 16 + p] = (p < 9)
            ? pack2h(wf[p * W_N + 2 * kk], wf[p * W_N + 2 * kk + 1]) : 0u;
    }
    // zero outputs for t >= train
    for (int idx = tid; idx < ROWS * T_N; idx += 256) {
        const int rr = idx / T_N, t = idx % T_N, rw = row0 + rr;
        if (t >= train && rw < Bn) out[rw * T_N + t] = 0.f;
    }
    __syncthreads();

    // ---- one vector-field evaluation: y (registers) -> Fbuf (LDS), 3 barriers
    auto vf_eval = [&](const float* yin) {
        // (1) H1 = softplus(w0 @ y + b0) -> Apack f16 pairs
        {
            const float y0 = yin[0], y1 = yin[1], y2 = yin[2];
            const int i0 = sg * 8;
#pragma unroll
            for (int u = 0; u < 8; u += 2) {
                const int ia = i0 + u, ib = i0 + u + 1;
                float a = sp_act(w0s[ia*3+0]*y0 + w0s[ia*3+1]*y1 + w0s[ia*3+2]*y2 + b0s[ia]);
                float b = sp_act(w0s[ib*3+0]*y0 + w0s[ib*3+1]*y1 + w0s[ib*3+2]*y2 + b0s[ib]);
                Apack[r * 64 + (ia >> 1)] = pack2h(a, b);
            }
        }
        __syncthreads();                                   // B1

        // (2) H2 = softplus(H1 @ w1^T + b1): 16x128x128, 8 waves x 4 WMMA
        {
            const int wave = tid >> 5;
            const int lane = tid & 31;
            const int hsel = lane >> 4;
            const int lrow = lane & 15;
            v8f acc = {};
#pragma unroll
            for (int ks = 0; ks < 4; ++ks) {
                union { v16h v; uint32_t u[8]; } af, bfv;
#pragma unroll
                for (int v = 0; v < 8; ++v) {
                    const int kk = ks * 16 + ((v < 4) ? 0 : 8) + hsel * 4 + (v & 3);
                    af.u[v]  = Apack[lrow * 64 + kk];
                    bfv.u[v] = Bpack[kk * W_N + wave * 16 + lrow];
                }
                acc = __builtin_amdgcn_wmma_f32_16x16x32_f16(
                        false, af.v, false, bfv.v, (short)0, acc, false, false);
            }
            union { v8f v; float f[8]; } au; au.v = acc;
            const int col = wave * 16 + lrow;
#pragma unroll
            for (int q = 0; q < 8; ++q) {
                h2h[(q + hsel * 8) * W_N + col] = (_Float16)sp_act(au.f[q] + sb1[col]);
            }
        }
        __syncthreads();                                   // B2

        // (3) F = tanh(H2 @ wf^T + bf): 16x16x128 on wave 0 (cols 9..15 dead)
        if (tid < 32) {
            const int hsel = tid >> 4;
            const int lrow = tid & 15;
            v8f acc = {};
#pragma unroll
            for (int ks = 0; ks < 4; ++ks) {
                union { v16h v; uint32_t u[8]; } af, bfv;
#pragma unroll
                for (int v = 0; v < 8; ++v) {
                    const int kk = ks * 16 + ((v < 4) ? 0 : 8) + hsel * 4 + (v & 3);
                    af.u[v]  = h2pack[lrow * 64 + kk];
                    bfv.u[v] = wfpack[kk * 16 + lrow];
                }
                acc = __builtin_amdgcn_wmma_f32_16x16x32_f16(
                        false, af.v, false, bfv.v, (short)0, acc, false, false);
            }
            union { v8f v; float f[8]; } au; au.v = acc;
            if (lrow < 9) {
#pragma unroll
                for (int q = 0; q < 8; ++q) {
                    Fbuf[q + hsel * 8][lrow] = tanh_fast(au.f[q] + bfs[lrow]);
                }
            }
        }
        __syncthreads();                                   // B3: Fbuf valid
    };

    // k = F(3x3) @ c  -- read per-thread from Fbuf (replicated across the row's 16 threads)
    auto getk = [&](float* k, const float* cc) {
#pragma unroll
        for (int a = 0; a < 3; ++a)
            k[a] = Fbuf[r][3*a+0]*cc[0] + Fbuf[r][3*a+1]*cc[1] + Fbuf[r][3*a+2]*cc[2];
    };

    // RK4 interval, state entirely in registers
    auto rk4_interval = [&](float dt, float* ybase, const float* cc) {
        const float h = dt * (1.f / NSUB);
        float ycur[3] = { ybase[0], ybase[1], ybase[2] };
        for (int sub = 0; sub < NSUB; ++sub) {
            float k[3], ksum[3];
            vf_eval(ycur); getk(k, cc);                              // k1
#pragma unroll
            for (int a = 0; a < 3; ++a) { ksum[a] = k[a]; ycur[a] = ybase[a] + 0.5f*h*k[a]; }
            vf_eval(ycur); getk(k, cc);                              // k2
#pragma unroll
            for (int a = 0; a < 3; ++a) { ksum[a] += 2.f*k[a]; ycur[a] = ybase[a] + 0.5f*h*k[a]; }
            vf_eval(ycur); getk(k, cc);                              // k3
#pragma unroll
            for (int a = 0; a < 3; ++a) { ksum[a] += 2.f*k[a]; ycur[a] = ybase[a] + h*k[a]; }
            vf_eval(ycur); getk(k, cc);                              // k4
#pragma unroll
            for (int a = 0; a < 3; ++a) {
                ybase[a] += (h/6.f)*(ksum[a] + k[a]);
                ycur[a]   = ybase[a];
            }
        }
    };

    // ---- init: y0 = (ts[0], ys[row][0], ts[0]*ys[row][0])
    float ybase[3];
    {
        float yv = rowok ? ys[row * T_N] : 0.f;
        ybase[0] = tss[0]; ybase[1] = yv; ybase[2] = tss[0] * yv;
        if (writer && rowok) out[row * T_N] = yv;
    }
    float prev2 = ybase[1];                  // channel-1 of all[s-2]

    // ---- Phase A: control segments (linear interpolation of observations)
    for (int seg = 0; seg < ctrl - 1; ++seg) {
        const float t0 = tss[seg], t1 = tss[seg + 1];
        const float dt = t1 - t0;
        const float ya = rowok ? ys[row * T_N + seg]     : 0.f;
        const float yb = rowok ? ys[row * T_N + seg + 1] : 0.f;
        float cc[3];
        cc[0] = 1.f;                          // (t1 - t0)/dt
        cc[1] = (yb - ya) / dt;
        cc[2] = (t1 * yb - t0 * ya) / dt;
        const float cur1 = ybase[1];
        rk4_interval(dt, ybase, cc);
        prev2 = cur1;
        if (writer && rowok) out[row * T_N + seg + 1] = ybase[1];
    }

    // ---- Phase B: autoregressive extrapolation steps
    for (int s = ctrl; s < train; ++s) {
        const float t0 = tss[s - 1], t1 = tss[s];
        const float dt = t1 - t0;
        const float a1 = ybase[1];            // all[s-1] channel 1
        float cc[3];
        cc[0] = 1.f;
        cc[1] = (a1 - prev2) / dt;
        cc[2] = ((2.f * a1 - prev2) * t1 - a1 * t0) / dt;
        rk4_interval(dt, ybase, cc);
        prev2 = a1;
        if (writer && rowok) out[row * T_N + s] = ybase[1];
    }
}

extern "C" void kernel_launch(void* const* d_in, const int* in_sizes, int n_in,
                              void* d_out, int out_size, void* d_ws, size_t ws_size,
                              hipStream_t stream) {
    const float* ts    = (const float*)d_in[0];
    const float* ys    = (const float*)d_in[1];
    const int*   ctrl  = (const int*)  d_in[2];
    /* d_in[3] = saveat (unused) */
    const int*   train = (const int*)  d_in[4];
    const float* w0    = (const float*)d_in[5];
    const float* b0    = (const float*)d_in[6];
    const float* w1    = (const float*)d_in[7];
    const float* b1    = (const float*)d_in[8];
    const float* wf    = (const float*)d_in[9];
    const float* bf    = (const float*)d_in[10];
    float* out = (float*)d_out;

    const int Bn   = in_sizes[1] / T_N;            // 256
    const int grid = (Bn + ROWS - 1) / ROWS;       // 16 workgroups
    cde_persistent_kernel<<<grid, 256, 0, stream>>>(ts, ys, ctrl, train,
                                                    w0, b0, w1, b1, wf, bf,
                                                    out, Bn);
}
